// CustomLSTM_44324062494922
// MI455X (gfx1250) — matrix operand order repaired and checked by reading, not compiled, not run on hardware
//
#include <hip/hip_runtime.h>
#include <hip/hip_bf16.h>
#include <stdint.h>

// ---------------------------------------------------------------------------
// LSTM for MI455X (gfx1250, wave32).
//   B=32, S=512, D=512, H=1024, 4H=4096
// Phase 1: gates_x[S,B,4096] = x @ [Wii|Wif|Wig|Wio] + bias  (bf16 WMMA, f32 acc,
//          1x4 N register blocking)
// Phase 2: 512 sequential step kernels:
//   h staged into LDS via TDM tensor_load_to_lds (TENSORcnt),
//   pre = gates_x[t] + h_{t-1} @ [Whi|Whf|Whg|Who]  (bf16 WMMA, A from ds_load)
//   i,f,o = sigmoid, g = tanh; c = f*c + i*g; h = o*tanh(c)
// ---------------------------------------------------------------------------

typedef __attribute__((ext_vector_type(16))) __bf16   v16bf;
typedef __attribute__((ext_vector_type(8)))  float    v8f;
typedef __attribute__((ext_vector_type(4)))  unsigned v4u;
typedef __attribute__((ext_vector_type(8)))  int      v8i;
typedef __attribute__((ext_vector_type(4)))  int      v4i;

union ABReg { v4u q[2]; v16bf v; };

__device__ __forceinline__ unsigned short f2bf(float f) {
  union { float f; unsigned u; } v; v.f = f;
  unsigned u = v.u;
  u += 0x7FFFu + ((u >> 16) & 1u);           // round-to-nearest-even
  return (unsigned short)(u >> 16);
}

__device__ __forceinline__ float sigmoidf_fast(float x) {
  return 1.0f / (1.0f + __expf(-x));
}
__device__ __forceinline__ float tanhf_fast(float x) {
  return 2.0f / (1.0f + __expf(-2.0f * x)) - 1.0f;
}

// Sizes
#define LB   32
#define LS   512
#define LD   512
#define LH   1024
#define LNG  4096   // 4*H

// ---------------------------------------------------------------------------
// TDM: async DMA of the 32x1024 bf16 h-matrix (row-major, stride 1024) into
// LDS at byte offset lds_off. D# built per CDNA5 ISA 8.3/8.4.
// ---------------------------------------------------------------------------
__device__ __forceinline__ void tdm_load_h_to_lds(const unsigned short* gptr,
                                                  unsigned lds_off) {
  unsigned long long ga = (unsigned long long)(uintptr_t)gptr;
  v4u g0;
  g0.x = 1u;                                        // count=1, user descriptor
  g0.y = lds_off;                                   // lds_addr (bytes)
  g0.z = (unsigned)(ga & 0xFFFFFFFFu);              // global_addr[31:0]
  g0.w = (unsigned)((ga >> 32) & 0x01FFFFFFu)       // global_addr[56:32]
       | (2u << 30);                                // type=2 ("image")
  v8i g1;
  g1[0] = (int)(1u << 16);        // wg_mask=0, data_size=1 (2 bytes)
  g1[1] = (int)(1024u << 16);     // tensor_dim0[15:0]=1024 (atomic_bar_addr=0)
  g1[2] = (int)(32u << 16);       // tensor_dim0 hi=0, tensor_dim1[15:0]=32
  g1[3] = (int)(1024u << 16);     // tensor_dim1 hi=0, tile_dim0=1024
  g1[4] = (int)32u;               // tile_dim1=32, tile_dim2=0
  g1[5] = (int)1024u;             // tensor_dim0_stride = 1024 (low 32)
  g1[6] = 0;                      // stride hi / tensor_dim1_stride lo
  g1[7] = 0;
  v4i z4 = {0, 0, 0, 0};
#if __clang_major__ >= 23
  v8i z8 = {0, 0, 0, 0, 0, 0, 0, 0};
  __builtin_amdgcn_tensor_load_to_lds(g0, g1, z4, z4, z8, 0);
#else
  __builtin_amdgcn_tensor_load_to_lds(g0, g1, z4, z4, 0);
#endif
}

// ---------------------------------------------------------------------------
// fp32 -> bf16 elementwise convert (x): 32*512*512 = 8388608 elems
// ---------------------------------------------------------------------------
__global__ __launch_bounds__(256) void k_convert_x(const float* __restrict__ src,
                                                   unsigned short* __restrict__ dst,
                                                   int n) {
  int i = blockIdx.x * 256 + threadIdx.x;
  if (i < n) dst[i] = f2bf(src[i]);
}

// ---------------------------------------------------------------------------
// Pack 4 weight matrices [Kdim,1024] into WMMA-B tiles over [Kdim, 4096].
// Tile (kt,nt) = 32x16 (KxN), 512 bf16; dest elem e:
//   tile = e/512, L = (e%512)/16, s = e%16
//   K = kt*32 + s + 16*(L>>4), N = nt*16 + (L&15)
// ---------------------------------------------------------------------------
__global__ __launch_bounds__(256) void k_pack_w(const float* __restrict__ W0,
                                                const float* __restrict__ W1,
                                                const float* __restrict__ W2,
                                                const float* __restrict__ W3,
                                                unsigned short* __restrict__ dst,
                                                int Kdim) {
  size_t e = (size_t)blockIdx.x * 256 + threadIdx.x;
  size_t total = (size_t)Kdim * LNG;
  if (e >= total) return;
  int within = (int)(e & 511);
  size_t tile = e >> 9;
  int L = within >> 4, s = within & 15;
  int kt = (int)(tile >> 8);
  int nt = (int)(tile & 255);
  int K = kt * 32 + s + 16 * (L >> 4);
  int N = nt * 16 + (L & 15);
  int gate = N >> 10, n = N & 1023;
  const float* W = (gate == 0) ? W0 : (gate == 1) ? W1 : (gate == 2) ? W2 : W3;
  dst[e] = f2bf(W[(size_t)K * LH + n]);
}

__global__ __launch_bounds__(256) void k_pack_bias(const float* __restrict__ b0,
                                                   const float* __restrict__ b1,
                                                   const float* __restrict__ b2,
                                                   const float* __restrict__ b3,
                                                   float* __restrict__ bias4) {
  int i = blockIdx.x * 256 + threadIdx.x;
  if (i >= LNG) return;
  int gate = i >> 10, n = i & 1023;
  const float* b = (gate == 0) ? b0 : (gate == 1) ? b1 : (gate == 2) ? b2 : b3;
  bias4[i] = b[n];
}

__global__ __launch_bounds__(256) void k_init_state(unsigned short* __restrict__ h_bf,
                                                    float* __restrict__ c_buf) {
  int i = blockIdx.x * 256 + threadIdx.x;
  if (i < LB * LH) { h_bf[i] = 0; c_buf[i] = 0.0f; }
}

// ---------------------------------------------------------------------------
// Input projections with 1x4 N-blocking: each wave computes 4 C-tiles sharing
// one A tile per K-step (2.5 b128 loads per WMMA instead of 4).
// wave-jobs: S * 2(mtiles) * 64(ngroups) = 65536 ; 8 waves/block -> 8192 blocks
// ---------------------------------------------------------------------------
__global__ __launch_bounds__(256) void k_gemm_x(const unsigned short* __restrict__ x_bf,
                                                const unsigned short* __restrict__ Wx_p,
                                                const float* __restrict__ bias4,
                                                float* __restrict__ gates) {
  const int w = threadIdx.x >> 5;
  const int L = threadIdx.x & 31;
  const size_t tid = (size_t)blockIdx.x * 8 + w;   // 0..65535
  const int t     = (int)(tid >> 7);               // /128
  const int rem   = (int)(tid & 127);
  const int mtile = rem >> 6;                      // 0..1
  const int ng    = rem & 63;                      // group of 4 n-tiles
  const int nt0   = ng << 2;
  const int col   = L & 15;
  const int half  = L >> 4;

  const int brow = mtile * 16 + col;               // batch row for A operand
  const unsigned short* ap = x_bf + ((size_t)brow * LS + t) * LD;

  v8f acc[4] = {{}, {}, {}, {}};
#pragma unroll 2
  for (int k0 = 0; k0 < LD; k0 += 32) {
    ABReg A;
    A.q[0] = *(const v4u*)(ap + k0 + 16 * half);
    A.q[1] = *(const v4u*)(ap + k0 + 16 * half + 8);
    const unsigned short* bt = Wx_p + ((size_t)(k0 >> 5) * 256 + nt0) * 512 + L * 16;
#pragma unroll
    for (int j = 0; j < 4; ++j) {
      ABReg Bm;
      Bm.q[0] = *(const v4u*)(bt + j * 512);
      Bm.q[1] = *(const v4u*)(bt + j * 512 + 8);
      acc[j] = __builtin_amdgcn_wmma_f32_16x16x32_bf16(false, A.v, false, Bm.v,
                                                       (short)0, acc[j], false, false);
    }
  }

  const int rbase = mtile * 16 + 8 * half;
  float* gr = gates + ((size_t)t * LB + rbase) * LNG + col;
#pragma unroll
  for (int j = 0; j < 4; ++j) {
    const int n0 = (nt0 + j) * 16;
    const float bb = bias4[n0 + col];
#pragma unroll
    for (int r = 0; r < 8; ++r) gr[(size_t)r * LNG + n0] = acc[j][r] + bb;
  }
}

// ---------------------------------------------------------------------------
// One recurrent step. 64 blocks x 256 threads; block owns a 16-wide hidden
// slice. h (32x1024 bf16, 64KB) is DMA'd into LDS once per block by the TDM;
// 8 waves (wave = mtile*4 + gate) each compute one 16x16 C tile with A from
// LDS, then gates are fused through the same LDS (overlaid after a barrier).
// ---------------------------------------------------------------------------
__global__ __launch_bounds__(256) void k_lstm_step(const unsigned short* __restrict__ Wh_p,
                                                   const float* __restrict__ gates,
                                                   unsigned short* __restrict__ h_bf,
                                                   float* __restrict__ c_buf,
                                                   float* __restrict__ out,
                                                   int t) {
  __shared__ alignas(16) unsigned short hs[LB * LH];   // 64 KB; reused for gate xfer

  const int w     = threadIdx.x >> 5;
  const int L     = threadIdx.x & 31;
  const int gate  = w & 3;
  const int mtile = w >> 2;
  const int nh0   = blockIdx.x * 16;          // hidden-unit slice
  const int col   = L & 15;
  const int half  = L >> 4;
  const int n0    = gate * LH + nh0;          // column base in [0,4096)

  // --- TDM: stage h into LDS (one wave issues; EXEC-independent DMA) ---
  if (threadIdx.x < 32) {
    tdm_load_h_to_lds(h_bf, (unsigned)(uintptr_t)hs);
    __builtin_amdgcn_s_wait_tensorcnt(0);
  }
  __syncthreads();

  const int arow = mtile * 16 + col;          // batch row for A operand
  const unsigned short* ap = hs + (size_t)arow * LH;
  const int ntcol = n0 >> 4;

  v8f acc = {};
#pragma unroll 4
  for (int k0 = 0; k0 < LH; k0 += 32) {
    ABReg A, Bm;
    A.q[0] = *(const v4u*)(ap + k0 + 16 * half);         // ds_load_b128
    A.q[1] = *(const v4u*)(ap + k0 + 16 * half + 8);
    const unsigned short* bt = Wh_p + ((size_t)(k0 >> 5) * 256 + ntcol) * 512 + L * 16;
    Bm.q[0] = *(const v4u*)(bt);
    Bm.q[1] = *(const v4u*)(bt + 8);
    acc = __builtin_amdgcn_wmma_f32_16x16x32_bf16(false, A.v, false, Bm.v,
                                                  (short)0, acc, false, false);
  }

  // add precomputed input projection
  const float* gx = gates + ((size_t)t * LB + mtile * 16 + 8 * half) * LNG + n0 + col;
#pragma unroll
  for (int r = 0; r < 8; ++r) acc[r] += gx[(size_t)r * LNG];

  // overlay the 8KB gate-exchange buffer on hs (h data is dead now)
  __syncthreads();
  float* pre = (float*)hs;                    // [8][16][16]
#pragma unroll
  for (int r = 0; r < 8; ++r) pre[(w << 8) + (r + 8 * half) * 16 + col] = acc[r];
  __syncthreads();

  // gate fusion: 512 (b,n) elements per block, 2 per thread
#pragma unroll
  for (int e = 0; e < 2; ++e) {
    const int flat = threadIdx.x + 256 * e;
    const int mt = flat >> 8;
    const int rr = (flat >> 4) & 15;
    const int cc = flat & 15;
    const int b = mt * 16 + rr;
    const int n = nh0 + cc;

    const float pi = pre[((mt * 4 + 0) << 8) + rr * 16 + cc];
    const float pf = pre[((mt * 4 + 1) << 8) + rr * 16 + cc];
    const float pg = pre[((mt * 4 + 2) << 8) + rr * 16 + cc];
    const float po = pre[((mt * 4 + 3) << 8) + rr * 16 + cc];

    const float ig = sigmoidf_fast(pi);
    const float fg = sigmoidf_fast(pf);
    const float gg = tanhf_fast(pg);
    const float og = sigmoidf_fast(po);

    const size_t sidx = (size_t)b * LH + n;
    const float cprev = c_buf[sidx];
    const float cnew  = fg * cprev + ig * gg;
    const float hnew  = og * tanhf_fast(cnew);

    c_buf[sidx] = cnew;
    h_bf[sidx]  = f2bf(hnew);

    const size_t oidx = ((size_t)b * LS + t) * LH + n;
    out[oidx] = hnew;                                   // outputs
    out[(size_t)LB * LS * LH + oidx] = cnew;            // cell_states
  }
}

// ---------------------------------------------------------------------------
// Host-side launcher
// ---------------------------------------------------------------------------
extern "C" void kernel_launch(void* const* d_in, const int* in_sizes, int n_in,
                              void* d_out, int out_size, void* d_ws, size_t ws_size,
                              hipStream_t stream) {
  const float* x    = (const float*)d_in[0];
  const float* W_ii = (const float*)d_in[1];
  const float* W_if = (const float*)d_in[2];
  const float* W_ig = (const float*)d_in[3];
  const float* W_io = (const float*)d_in[4];
  const float* W_hi = (const float*)d_in[5];
  const float* W_hf = (const float*)d_in[6];
  const float* W_hg = (const float*)d_in[7];
  const float* W_ho = (const float*)d_in[8];
  const float* b_i  = (const float*)d_in[9];
  const float* b_f  = (const float*)d_in[10];
  const float* b_g  = (const float*)d_in[11];
  const float* b_o  = (const float*)d_in[12];
  float* out = (float*)d_out;

  // Workspace carve-up (bf16 region first, then fp32 region; all 16B aligned)
  unsigned short* u16b = (unsigned short*)d_ws;
  unsigned short* Wx_p = u16b;                               //  512*4096
  unsigned short* Wh_p = Wx_p + (size_t)LD * LNG;            // 1024*4096
  unsigned short* x_bf = Wh_p + (size_t)LH * LNG;            // 32*512*512
  unsigned short* h_bf = x_bf + (size_t)LB * LS * LD;        // 32*1024
  float* fbase = (float*)(h_bf + (size_t)LB * LH);
  float* bias4 = fbase;                                      // 4096
  float* c_buf = bias4 + LNG;                                // 32*1024
  float* gates = c_buf + (size_t)LB * LH;                    // 512*32*4096

  // Phase 0: conversion / packing / state init
  {
    const int nx = LB * LS * LD;                             // 8388608
    k_convert_x<<<dim3(nx / 256), dim3(256), 0, stream>>>(x, x_bf, nx);

    const size_t wx_total = (size_t)LD * LNG;                // 2097152
    k_pack_w<<<dim3((unsigned)(wx_total / 256)), dim3(256), 0, stream>>>(
        W_ii, W_if, W_ig, W_io, Wx_p, LD);

    const size_t wh_total = (size_t)LH * LNG;                // 4194304
    k_pack_w<<<dim3((unsigned)(wh_total / 256)), dim3(256), 0, stream>>>(
        W_hi, W_hf, W_hg, W_ho, Wh_p, LH);

    k_pack_bias<<<dim3(LNG / 256), dim3(256), 0, stream>>>(b_i, b_f, b_g, b_o, bias4);
    k_init_state<<<dim3((LB * LH) / 256), dim3(256), 0, stream>>>(h_bf, c_buf);
  }

  // Phase 1: all input-to-hidden projections (one big WMMA GEMM)
  {
    // wave-jobs = S * 2 * 64 = 65536 ; 8 waves per block
    k_gemm_x<<<dim3(65536 / 8), dim3(256), 0, stream>>>(x_bf, Wx_p, bias4, gates);
  }

  // Phase 2: sequential recurrence, one launch per time step
  for (int t = 0; t < LS; ++t) {
    k_lstm_step<<<dim3(LH / 16), dim3(256), 0, stream>>>(Wh_p, gates, h_bf, c_buf, out, t);
  }
}
